// EnhancedMamba2Block_44117904065245
// MI455X (gfx1250) — compile-verified, matrix operand order checked by reading, not run on hardware
//
#include <hip/hip_runtime.h>
#include <hip/hip_bf16.h>
#include <math.h>

#ifndef __has_builtin
#define __has_builtin(x) 0
#endif

// ---------------- problem constants ----------------
#define BATCH 2
#define SEQ   4096
#define DM    1024      // d_model
#define DS    16        // d_state
#define DCONV 4
#define NCH   64        // scan chunks
#define LC    (SEQ / NCH)

// ---------------- types ----------------
typedef __attribute__((ext_vector_type(16))) __bf16 v16bf;
typedef __attribute__((ext_vector_type(2)))  __bf16 v2bf;
typedef __attribute__((ext_vector_type(8)))  float  v8f;

// exact parameter types for the async-LDS builtin (from hipcc diagnostic):
// pointer to 16-byte int vector, AS1 (global) source / AS3 (LDS) destination
typedef int v4i_vs __attribute__((__vector_size__(16)));
typedef __attribute__((address_space(1))) v4i_vs g_v4i;
typedef __attribute__((address_space(3))) v4i_vs l_v4i;

#if __has_builtin(__builtin_amdgcn_global_load_async_to_lds_b128)
#define HAVE_ASYNC_LDS 1
#else
#define HAVE_ASYNC_LDS 0
#endif

__device__ __forceinline__ __bf16 f2bf(float f) {
  union { float f; unsigned u; } v; v.f = f;
  unsigned u = v.u + 0x7FFFu + ((v.u >> 16) & 1u);   // round-to-nearest-even
  unsigned short s = (unsigned short)(u >> 16);
  __bf16 b;
  __builtin_memcpy(&b, &s, sizeof(b));
  return b;
}

__device__ __forceinline__ v2bf pk_bf16(float a, float b) {
#if __has_builtin(__builtin_amdgcn_cvt_pk_bf16_f32)
  return __builtin_amdgcn_cvt_pk_bf16_f32(a, b);
#else
  v2bf r; r[0] = f2bf(a); r[1] = f2bf(b); return r;
#endif
}

__device__ __forceinline__ float sigmoidf_(float x) { return 1.f / (1.f + __expf(-x)); }
__device__ __forceinline__ float siluf_(float x)    { return x * sigmoidf_(x); }
__device__ __forceinline__ float softplusf_(float x){ return (x > 20.f) ? x : log1pf(__expf(x)); }

// 16-byte global -> LDS stage; async path (ASYNCcnt-tracked) when available.
__device__ __forceinline__ void stage16(const __bf16* g, __bf16* l) {
#if HAVE_ASYNC_LDS
  __builtin_amdgcn_global_load_async_to_lds_b128(
      (g_v4i*)(size_t)g, (l_v4i*)(unsigned)(size_t)l, 0, 0);
#else
  *(uint4*)l = *(const uint4*)g;
#endif
}
__device__ __forceinline__ void wait_async_le3() {
#if HAVE_ASYNC_LDS
# if __has_builtin(__builtin_amdgcn_s_wait_asynccnt)
  __builtin_amdgcn_s_wait_asynccnt(3);
# else
  asm volatile("s_wait_asynccnt 0x3" ::: "memory");
# endif
#endif
}
__device__ __forceinline__ void wait_async_0() {
#if HAVE_ASYNC_LDS
# if __has_builtin(__builtin_amdgcn_s_wait_asynccnt)
  __builtin_amdgcn_s_wait_asynccnt(0);
# else
  asm volatile("s_wait_asynccnt 0x0" ::: "memory");
# endif
#endif
}

// ---------------- fp32 -> bf16 convert (with zero padding) ----------------
__global__ __launch_bounds__(256)
void cvt_f32_bf16(const float* __restrict__ src, __bf16* __restrict__ dst,
                  long long srcN, long long dstN)
{
  long long i = ((long long)blockIdx.x * 256 + threadIdx.x) * 4;
  if (i >= dstN) return;
  float4 v = make_float4(0.f, 0.f, 0.f, 0.f);
  if (i < srcN) v = *(const float4*)(src + i);
  union { v2bf b; unsigned u; } lo, hi;
  lo.b = pk_bf16(v.x, v.y);
  hi.b = pk_bf16(v.z, v.w);
  uint2 o; o.x = lo.u; o.y = hi.u;
  *(uint2*)(dst + i) = o;
}

// ---------------- WMMA GEMM: C[M,N] = epi(A[M,K] @ W[N,K]^T + bias) ----------------
// A, W are bf16 row-major (leading dim K). Block tile 128x64, K-step 32,
// 256 threads = 8 waves, each wave a 32x32 tile (2x2 WMMA 16x16x32).
// LDS ping-pong buffers filled with global_load_async_to_lds_b128.
#define BM 128
#define BN 64
#define BK 32
#define LDT 40   // LDS row stride in bf16 (80B: 16B-aligned chunks, spreads banks)

__global__ __launch_bounds__(256)
void wmma_gemm_bf16(const __bf16* __restrict__ A, const __bf16* __restrict__ W,
                    const float* __restrict__ bias, float* __restrict__ C,
                    int M, int N, int K, int epi /*0=none,1=softplus*/)
{
  __shared__ __align__(16) __bf16 As[2][BM * LDT];
  __shared__ __align__(16) __bf16 Ws[2][BN * LDT];

  const int tid   = threadIdx.x;
  const int lane  = tid & 31;
  const int wid   = tid >> 5;
  const int mBase = blockIdx.y * BM;
  const int nBase = blockIdx.x * BN;
  const int wm    = (wid & 3) * 32;
  const int wn    = (wid >> 2) * 32;

  const v8f vzero = {0.f,0.f,0.f,0.f,0.f,0.f,0.f,0.f};
  v8f acc[2][2];
  acc[0][0] = vzero; acc[0][1] = vzero; acc[1][0] = vzero; acc[1][1] = vzero;

  const int kb = (lane >> 4) << 3;   // fragment K-base for this lane

  // stage one BMxBK + BNxBK tile pair (16B chunks; 2 A-chunks + 1 W-chunk per thread)
  auto stage_step = [&](int k0, int buf) {
    #pragma unroll
    for (int q = 0; q < 2; ++q) {
      int ca = tid * 2 + q;
      int r = ca >> 2, c = ca & 3;
      stage16(A + (size_t)(mBase + r) * (size_t)K + k0 + c * 8,
              &As[buf][r * LDT + c * 8]);
    }
    {
      int r = tid >> 2, c = tid & 3;
      stage16(W + (size_t)(nBase + r) * (size_t)K + k0 + c * 8,
              &Ws[buf][r * LDT + c * 8]);
    }
  };

  const int nsteps = K / BK;
  stage_step(0, 0);
  for (int s = 0; s < nsteps; ++s) {
    const int cur = s & 1;
    if (s + 1 < nsteps) { stage_step((s + 1) * BK, cur ^ 1); wait_async_le3(); }
    else                { wait_async_0(); }
    __syncthreads();

    // fragments per ISA 16-bit 16x32 layout: row = lane%16, k = kb + (h&7) + ((h&8)<<1)
    v16bf af[2], bfv[2];
    #pragma unroll
    for (int i = 0; i < 2; ++i) {
      const __bf16* rp = &As[cur][(wm + i * 16 + (lane & 15)) * LDT];
      #pragma unroll
      for (int h = 0; h < 16; ++h) af[i][h] = rp[kb + (h & 7) + ((h & 8) << 1)];
    }
    #pragma unroll
    for (int j = 0; j < 2; ++j) {
      const __bf16* rp = &Ws[cur][(wn + j * 16 + (lane & 15)) * LDT];
      #pragma unroll
      for (int h = 0; h < 16; ++h) bfv[j][h] = rp[kb + (h & 7) + ((h & 8) << 1)];
    }
    #pragma unroll
    for (int i = 0; i < 2; ++i)
      #pragma unroll
      for (int j = 0; j < 2; ++j)
        acc[i][j] = __builtin_amdgcn_wmma_f32_16x16x32_bf16(
            false, af[i], false, bfv[j], (short)0, acc[i][j], false, false);
    __syncthreads();
  }

  // C/D layout: VGPR r -> row r (lanes 0-15) / r+8 (lanes 16-31); col = lane%16
  const int rowOff = (lane >> 4) << 3;
  const int col    = lane & 15;
  #pragma unroll
  for (int i = 0; i < 2; ++i) {
    #pragma unroll
    for (int j = 0; j < 2; ++j) {
      int cGlob = nBase + wn + j * 16 + col;
      if (cGlob < N) {
        float bv = bias ? bias[cGlob] : 0.f;
        #pragma unroll
        for (int r = 0; r < 8; ++r) {
          int rGlob = mBase + wm + i * 16 + rowOff + r;
          float v = acc[i][j][r] + bv;
          if (epi == 1) v = softplusf_(v);
          C[(size_t)rGlob * N + cGlob] = v;
        }
      }
    }
  }
}

// ---------------- causal depthwise conv(4) + bias + SiLU ----------------
__global__ __launch_bounds__(256)
void conv_silu(const float* __restrict__ xz, const float* __restrict__ wconv,
               const float* __restrict__ bconv, float* __restrict__ xssm,
               __bf16* __restrict__ xssm_bf)
{
  size_t idx = (size_t)blockIdx.x * 256 + threadIdx.x;   // (b*SEQ + t)*DM + d
  int d = (int)(idx % DM);
  size_t bt = idx / DM;
  int t = (int)(bt % SEQ);
  int b = (int)(bt / SEQ);
  float acc = bconv[d];
  #pragma unroll
  for (int j = 0; j < DCONV; ++j) {
    int tt = t - (DCONV - 1) + j;
    if (tt >= 0)
      acc += xz[((size_t)b * SEQ + tt) * (2 * DM) + d] * wconv[d * DCONV + j];
  }
  float r = siluf_(acc);
  xssm[idx]    = r;
  xssm_bf[idx] = f2bf(r);
}

// ---------------- selective scan, 3-phase chunk decomposition ----------------
__global__ __launch_bounds__(256)
void scan_phase1(const float* __restrict__ dtb, const float* __restrict__ bcb,
                 const float* __restrict__ A_log,
                 float* __restrict__ P, float* __restrict__ S)
{
  const int b = blockIdx.x / NCH;
  const int c = blockIdx.x % NCH;
  const int d = blockIdx.y * 256 + threadIdx.x;
  __shared__ float Bs[LC * DS];
  const float* bcp = bcb + ((size_t)b * SEQ + (size_t)c * LC) * (2 * DS);
  for (int i = threadIdx.x; i < LC * DS; i += 256)
    Bs[i] = bcp[(i >> 4) * (2 * DS) + (i & 15)];
  __syncthreads();

  float An[DS], st[DS];
  #pragma unroll
  for (int n = 0; n < DS; ++n) { An[n] = -__expf(A_log[d * DS + n]); st[n] = 0.f; }
  float sdt = 0.f;
  const float* dtp = dtb + ((size_t)b * SEQ + (size_t)c * LC) * DM + d;
  for (int t = 0; t < LC; ++t) {
    float dtv = dtp[(size_t)t * DM];
    sdt += dtv;
    #pragma unroll
    for (int n = 0; n < DS; ++n)
      st[n] = st[n] * __expf(dtv * An[n]) + Bs[t * DS + n];
  }
  size_t o = (((size_t)b * NCH + c) * DM + d) * DS;
  #pragma unroll
  for (int n = 0; n < DS; ++n) { S[o + n] = st[n]; P[o + n] = __expf(sdt * An[n]); }
}

__global__ __launch_bounds__(256)
void scan_phase2(const float* __restrict__ P, const float* __restrict__ S,
                 const float* __restrict__ state0,
                 float* __restrict__ Sin, float* __restrict__ state_out)
{
  int g = blockIdx.x * 256 + threadIdx.x;        // (b*DM + d)*DS + n
  int b = g / (DM * DS);
  int rem = g - b * (DM * DS);
  float carry = state0[g];
  for (int c = 0; c < NCH; ++c) {
    size_t o = ((size_t)b * NCH + c) * (size_t)(DM * DS) + rem;
    float nxt = P[o] * carry + S[o];
    Sin[o] = carry;
    carry = nxt;
  }
  state_out[g] = carry;
}

__global__ __launch_bounds__(256)
void scan_phase3(const float* __restrict__ dtb, const float* __restrict__ bcb,
                 const float* __restrict__ A_log, const float* __restrict__ Sin,
                 const float* __restrict__ xz, const float* __restrict__ Dskip,
                 const float* __restrict__ xssm, __bf16* __restrict__ y_bf)
{
  const int b = blockIdx.x / NCH;
  const int c = blockIdx.x % NCH;
  const int d = blockIdx.y * 256 + threadIdx.x;
  __shared__ float Bs[LC * DS];
  __shared__ float Cs[LC * DS];
  const float* bcp = bcb + ((size_t)b * SEQ + (size_t)c * LC) * (2 * DS);
  for (int i = threadIdx.x; i < LC * DS; i += 256) {
    Bs[i] = bcp[(i >> 4) * (2 * DS) + (i & 15)];
    Cs[i] = bcp[(i >> 4) * (2 * DS) + DS + (i & 15)];
  }
  __syncthreads();

  float An[DS], st[DS];
  size_t o = (((size_t)b * NCH + c) * DM + d) * DS;
  #pragma unroll
  for (int n = 0; n < DS; ++n) { An[n] = -__expf(A_log[d * DS + n]); st[n] = Sin[o + n]; }
  const float Dv = Dskip[d];
  const float* dtp = dtb + ((size_t)b * SEQ + (size_t)c * LC) * DM + d;
  for (int t = 0; t < LC; ++t) {
    float dtv = dtp[(size_t)t * DM];
    float y = 0.f;
    #pragma unroll
    for (int n = 0; n < DS; ++n) {
      st[n] = st[n] * __expf(dtv * An[n]) + Bs[t * DS + n];
      y += st[n] * Cs[t * DS + n];
    }
    size_t gi = ((size_t)b * SEQ + (size_t)c * LC + t) * DM + d;
    float xv = xssm[gi];
    float zv = xz[((size_t)b * SEQ + (size_t)c * LC + t) * (2 * DM) + DM + d];
    y_bf[gi] = f2bf((y + xv * Dv) * siluf_(zv));
  }
}

// ---------------- launcher ----------------
static inline void* bump(char*& p, size_t bytes) {
  void* r = (void*)p;
  p += (bytes + 255) & ~(size_t)255;
  return r;
}

extern "C" void kernel_launch(void* const* d_in, const int* in_sizes, int n_in,
                              void* d_out, int out_size, void* d_ws, size_t ws_size,
                              hipStream_t stream) {
  const float* x      = (const float*)d_in[0];
  const float* state0 = (const float*)d_in[1];
  const float* w_in   = (const float*)d_in[2];
  const float* w_conv = (const float*)d_in[3];
  const float* b_conv = (const float*)d_in[4];
  const float* w_x    = (const float*)d_in[5];
  const float* w_dt   = (const float*)d_in[6];
  const float* b_dt   = (const float*)d_in[7];
  const float* A_log  = (const float*)d_in[8];
  const float* Dsk    = (const float*)d_in[9];
  const float* w_out  = (const float*)d_in[10];

  float* out       = (float*)d_out;                       // (B,SEQ,DM)
  float* state_out = out + (size_t)BATCH * SEQ * DM;      // (B,DM,DS)

  const size_t MR = (size_t)BATCH * SEQ;                  // 8192 rows
  char* p = (char*)d_ws;
  float*  xz      = (float*) bump(p, MR * 2 * DM * 4);
  float*  xssm    = (float*) bump(p, MR * DM * 4);
  float*  dtb     = (float*) bump(p, MR * DM * 4);
  float*  bcb     = (float*) bump(p, MR * 2 * DS * 4);
  float*  P       = (float*) bump(p, (size_t)BATCH * NCH * DM * DS * 4);
  float*  S       = (float*) bump(p, (size_t)BATCH * NCH * DM * DS * 4);
  float*  Sin     = (float*) bump(p, (size_t)BATCH * NCH * DM * DS * 4);
  __bf16* x_bf    = (__bf16*)bump(p, MR * DM * 2);        // reused as y_bf
  __bf16* w_in_bf = (__bf16*)bump(p, (size_t)2 * DM * DM * 2);
  __bf16* w_dt_bf = (__bf16*)bump(p, (size_t)DM * DM * 2);
  __bf16* w_x_bf  = (__bf16*)bump(p, (size_t)BN * DM * 2);   // padded 32->64 rows
  __bf16* w_out_bf= (__bf16*)bump(p, (size_t)DM * DM * 2);
  __bf16* xssm_bf = (__bf16*)bump(p, MR * DM * 2);
  __bf16* y_bf    = x_bf;                                  // x_bf dead after GEMM-1

  dim3 blk(256);
  // one-shot fp32->bf16 conversion (bandwidth-bound, removes VALU from GEMMs)
  cvt_f32_bf16<<<(unsigned)(MR * DM / 1024), blk, 0, stream>>>(x, x_bf, MR * DM, MR * DM);
  cvt_f32_bf16<<<(unsigned)(2 * DM * DM / 1024), blk, 0, stream>>>(w_in, w_in_bf,
      (long long)2 * DM * DM, (long long)2 * DM * DM);
  cvt_f32_bf16<<<(unsigned)(DM * DM / 1024), blk, 0, stream>>>(w_dt, w_dt_bf,
      (long long)DM * DM, (long long)DM * DM);
  cvt_f32_bf16<<<(unsigned)(BN * DM / 1024), blk, 0, stream>>>(w_x, w_x_bf,
      (long long)2 * DS * DM, (long long)BN * DM);          // zero-pad rows 32..63
  cvt_f32_bf16<<<(unsigned)(DM * DM / 1024), blk, 0, stream>>>(w_out, w_out_bf,
      (long long)DM * DM, (long long)DM * DM);

  // 1) xz = x @ w_in^T
  wmma_gemm_bf16<<<dim3((2 * DM) / BN, MR / BM), blk, 0, stream>>>(
      x_bf, w_in_bf, nullptr, xz, (int)MR, 2 * DM, DM, 0);
  // 2) depthwise causal conv + bias + SiLU
  conv_silu<<<(unsigned)(MR * DM / 256), blk, 0, stream>>>(xz, w_conv, b_conv, xssm, xssm_bf);
  // 3) dt = softplus(xssm @ w_dt^T + b_dt)
  wmma_gemm_bf16<<<dim3(DM / BN, MR / BM), blk, 0, stream>>>(
      xssm_bf, w_dt_bf, b_dt, dtb, (int)MR, DM, DM, 1);
  // 4) BC = xssm @ w_x^T  (N = 32; weights zero-padded to the 64-col tile)
  wmma_gemm_bf16<<<dim3(1, MR / BM), blk, 0, stream>>>(
      xssm_bf, w_x_bf, nullptr, bcb, (int)MR, 2 * DS, DM, 0);
  // 5-7) chunked selective scan
  scan_phase1<<<dim3(BATCH * NCH, DM / 256), blk, 0, stream>>>(dtb, bcb, A_log, P, S);
  scan_phase2<<<(BATCH * DM * DS) / 256, blk, 0, stream>>>(P, S, state0, Sin, state_out);
  scan_phase3<<<dim3(BATCH * NCH, DM / 256), blk, 0, stream>>>(
      dtb, bcb, A_log, Sin, xz, Dsk, xssm, y_bf);
  // 8) out = y @ w_out^T
  wmma_gemm_bf16<<<dim3(DM / BN, MR / BM), blk, 0, stream>>>(
      y_bf, w_out_bf, nullptr, out, (int)MR, DM, DM, 0);
}
